// GraphConvBnRelu_45655502356536
// MI455X (gfx1250) — matrix operand order repaired and checked by reading, not compiled, not run on hardware
//
#include <hip/hip_runtime.h>

typedef __attribute__((ext_vector_type(16))) _Float16 v16h;
typedef __attribute__((ext_vector_type(8)))  float    v8f;

#define C_CH   128
#define CPG    32      // channels per GroupNorm group
#define NGRP   4
#define TILE_M 16
#define NTILES 8       // C_OUT / 16
#define KCHUNK 4       // C_IN / 32

// ---------------- zero accumulators ----------------
__global__ void gc_zero_kernel(float* __restrict__ agg, float* __restrict__ deg,
                               int n_agg, int n_deg) {
    int i = blockIdx.x * blockDim.x + threadIdx.x;
    int stride = gridDim.x * blockDim.x;
    for (int j = i; j < n_agg; j += stride) agg[j] = 0.0f;
    for (int j = i; j < n_deg; j += stride) deg[j] = 0.0f;
}

// ---------------- convert + swizzle W into per-lane WMMA B fragments ----------------
// Whs[((ntile*KCHUNK + kc)*32 + lane)*16 + h] = W[k][col]  (f16)
//   where col = ntile*16 + (lane&15), k = kc*32 + (lane>=16)*16 + h
// so each lane's 16-half B fragment is 32 contiguous bytes.
__global__ void gc_convw_kernel(const float* __restrict__ W, _Float16* __restrict__ Whs) {
    int i = blockIdx.x * blockDim.x + threadIdx.x;
    if (i >= NTILES * KCHUNK * 32 * 16) return;
    int h     = i & 15;
    int lane  = (i >> 4) & 31;
    int kc    = (i >> 9) & (KCHUNK - 1);
    int ntile = i >> 11;
    int col   = ntile * 16 + (lane & 15);
    int k     = kc * 32 + ((lane >= 16) ? 16 : 0) + h;
    Whs[i] = (_Float16)W[k * C_CH + col];
}

// ---------------- edge scatter: one wave (32 lanes) per edge ----------------
__global__ void gc_scatter_kernel(const float* __restrict__ data,
                                  const int* __restrict__ ei,   // [2,E]
                                  float* __restrict__ agg,
                                  float* __restrict__ deg, int E) {
    int gid  = blockIdx.x * blockDim.x + threadIdx.x;
    int lane = gid & 31;
    int edge = gid >> 5;
    if (edge >= E) return;
    int src = ei[edge];
    int dst = ei[E + edge];
    const float4 v = ((const float4*)(data + (long long)src * C_CH))[lane];
    float* ap = agg + (long long)dst * C_CH + lane * 4;
    unsafeAtomicAdd(ap + 0, v.x);
    unsafeAtomicAdd(ap + 1, v.y);
    unsafeAtomicAdd(ap + 2, v.z);
    unsafeAtomicAdd(ap + 3, v.w);
    if (lane == 0) unsafeAtomicAdd(deg + dst, 1.0f);
}

// ---------------- fused: mean -> f16 WMMA GEMM -> bias -> GroupNorm -> ReLU ----------------
__global__ void __launch_bounds__(256)
gc_gemm_gn_relu_kernel(float* __restrict__ out,            // agg in, final result out
                       const float* __restrict__ deg,
                       const _Float16* __restrict__ Whs,   // pre-swizzled B fragments
                       const float* __restrict__ bias,
                       const float* __restrict__ gamma,
                       const float* __restrict__ beta,
                       int N) {
    __shared__ _Float16 a_lds[TILE_M][C_CH];   // 4 KB
    __shared__ float    o_lds[TILE_M][C_CH];   // 8 KB
    __shared__ float    mu_s[TILE_M][NGRP];
    __shared__ float    rs_s[TILE_M][NGRP];

    const int tid  = threadIdx.x;
    const int base = blockIdx.x * TILE_M;

    // Phase 1: load agg tile as float4, divide by max(deg,1), convert to f16
    for (int idx = tid; idx < TILE_M * (C_CH / 4); idx += 256) {
        int r  = idx >> 5;               // node in tile
        int c4 = idx & 31;               // float4 column
        int row = base + r;
        float4 v = make_float4(0.f, 0.f, 0.f, 0.f);
        if (row < N) {
            float d = fmaxf(deg[row], 1.0f);
            float4 a = ((const float4*)(out + (long long)row * C_CH))[c4];
            float inv = 1.0f / d;
            v = make_float4(a.x * inv, a.y * inv, a.z * inv, a.w * inv);
        }
        _Float16* dst = &a_lds[r][c4 * 4];
        dst[0] = (_Float16)v.x;
        dst[1] = (_Float16)v.y;
        dst[2] = (_Float16)v.z;
        dst[3] = (_Float16)v.w;
    }
    __syncthreads();

    // Phase 2: WMMA  D[16 nodes x 16 cols] += A[16x32] * B[32x16], K = 128
    const int wave = tid >> 5;
    const int lane = tid & 31;
    const int half = (lane < 16) ? 0 : 1;
    const int mrow = lane & 15;
    const int ncol = wave * 16 + (lane & 15);

    v8f acc = {};
#pragma unroll
    for (int kc = 0; kc < KCHUNK; ++kc) {
        const int kb = kc * 32;
        // A fragment: halves 0-7 = K kb+half*16.., halves 8-15 = K kb+half*16+8..
        // both are 8 consecutive halves -> two 16B LDS reads
        v16h afrag;
#pragma unroll
        for (int vv = 0; vv < 8; ++vv) {
            afrag[vv]     = a_lds[mrow][kb + half * 16 + vv];
            afrag[vv + 8] = a_lds[mrow][kb + half * 16 + 8 + vv];
        }
        // B fragment: 32 contiguous bytes from pre-swizzled Whs
        const v16h bfrag =
            *(const v16h*)&Whs[(((wave * KCHUNK) + kc) * 32 + lane) * 16];
        acc = __builtin_amdgcn_wmma_f32_16x16x32_f16(
            false, afrag, false, bfrag, (short)0, acc, false, false);
    }

    // bias + stage to LDS; C/D layout: acc[i] -> node = i + 8*half, col = ncol
    float bv = bias[ncol];
#pragma unroll
    for (int i = 0; i < 8; ++i) {
        o_lds[i + half * 8][ncol] = acc[i] + bv;
    }
    __syncthreads();

    // Phase 3: GroupNorm stats, one thread per (node, group)
    if (tid < TILE_M * NGRP) {
        int node = tid >> 2;
        int g    = tid & 3;
        float s = 0.0f, ss = 0.0f;
#pragma unroll
        for (int c = 0; c < CPG; ++c) {
            float x = o_lds[node][g * CPG + c];
            s  += x;
            ss += x * x;
        }
        float mu  = s * (1.0f / CPG);
        float var = ss * (1.0f / CPG) - mu * mu;
        mu_s[node][g] = mu;
        rs_s[node][g] = rsqrtf(var + 1e-5f);
    }
    __syncthreads();

    // Phase 4: normalize + affine + ReLU + vectorized store
    for (int idx = tid; idx < TILE_M * (C_CH / 4); idx += 256) {
        int r  = idx >> 5;
        int c4 = idx & 31;
        int row = base + r;
        if (row < N) {
            int c = c4 * 4;
            int g = c >> 5;
            float mu = mu_s[r][g];
            float rs = rs_s[r][g];
            float4 y;
            y.x = fmaxf((o_lds[r][c + 0] - mu) * rs * gamma[c + 0] + beta[c + 0], 0.f);
            y.y = fmaxf((o_lds[r][c + 1] - mu) * rs * gamma[c + 1] + beta[c + 1], 0.f);
            y.z = fmaxf((o_lds[r][c + 2] - mu) * rs * gamma[c + 2] + beta[c + 2], 0.f);
            y.w = fmaxf((o_lds[r][c + 3] - mu) * rs * gamma[c + 3] + beta[c + 3], 0.f);
            ((float4*)(out + (long long)row * C_CH))[c4] = y;
        }
    }
}

extern "C" void kernel_launch(void* const* d_in, const int* in_sizes, int n_in,
                              void* d_out, int out_size, void* d_ws, size_t ws_size,
                              hipStream_t stream) {
    const float* data  = (const float*)d_in[0];
    const float* W     = (const float*)d_in[1];
    const float* bias  = (const float*)d_in[2];
    const float* gamma = (const float*)d_in[3];
    const float* beta  = (const float*)d_in[4];
    const int*   ei    = (const int*)d_in[5];
    // d_in[6] = depth (unused)

    const int N = in_sizes[0] / C_CH;
    const int E = in_sizes[5] / 2;

    float* out = (float*)d_out;                 // doubles as agg accumulator
    float* deg = (float*)d_ws;
    size_t deg_bytes = (((size_t)N * sizeof(float)) + 255) & ~(size_t)255;
    _Float16* Whs = (_Float16*)((char*)d_ws + deg_bytes);

    gc_zero_kernel<<<2048, 256, 0, stream>>>(out, deg, N * C_CH, N);

    gc_convw_kernel<<<(C_CH * C_CH + 255) / 256, 256, 0, stream>>>(W, Whs);

    long long sthreads = (long long)E * 32;
    int sblocks = (int)((sthreads + 255) / 256);
    gc_scatter_kernel<<<sblocks, 256, 0, stream>>>(data, ei, out, deg, E);

    int gblocks = (N + TILE_M - 1) / TILE_M;
    gc_gemm_gn_relu_kernel<<<gblocks, 256, 0, stream>>>(out, deg, Whs, bias, gamma, beta, N);
}
